// CrossChannelInterp_30623116820684
// MI455X (gfx1250) — matrix operand order repaired and checked by reading, not compiled, not run on hardware
//
#include <hip/hip_runtime.h>
#include <hip/hip_bf16.h>

// Problem constants
#define B_   32
#define D_   512
#define T_   2048
#define D3_  1536

typedef __attribute__((ext_vector_type(16))) __bf16 v16bf;
typedef __attribute__((ext_vector_type(8)))  __bf16 v8bf;
typedef __attribute__((ext_vector_type(8)))  float  v8f;
typedef __attribute__((ext_vector_type(4)))  float  v4f;

// ---------------------------------------------------------------------------
// Kernel 0: Wt[n][k] = bf16(W[k][n])  (512x512, tiny — 1 MB read)
// ---------------------------------------------------------------------------
__global__ void k_wt(const float* __restrict__ W, __bf16* __restrict__ Wt) {
  int idx = blockIdx.x * 256 + threadIdx.x;    // 0..262143
  int k = idx & 511;
  int n = idx >> 9;
  Wt[(size_t)n * 512 + k] = (__bf16)W[(size_t)k * 512 + n];
}

// ---------------------------------------------------------------------------
// Kernel 1: mean[b][d] = (1/T) sum_t x[b][d][t]   (one wave32 per row)
// ---------------------------------------------------------------------------
__global__ void k_mean(const float* __restrict__ x, float* __restrict__ mean) {
  int wid  = blockIdx.x * 8 + (threadIdx.x >> 5);   // 0..16383 = b*512+d
  int lane = threadIdx.x & 31;
  int b = wid >> 9;
  int d = wid & 511;
  const float* row = x + ((size_t)b * D3_ + d) * T_;
  float s = 0.f;
  for (int t = lane; t < T_; t += 32) s += row[t];
#pragma unroll
  for (int off = 16; off >= 1; off >>= 1) s += __shfl_xor(s, off, 32);
  if (lane == 0) mean[wid] = s * (1.0f / T_);
}

// ---------------------------------------------------------------------------
// Kernel 2: per-(b,t) channel softmax of w; intensity = exp(w) out;
//           A[b][t][d] = bf16( sm * (y - mean) )   (d-contiguous for GEMM)
// Block: 256 threads handles 512(d) x 16(t) tile. LDS = 2 x 32KB.
// ---------------------------------------------------------------------------
__global__ void k_softmax_A(const float* __restrict__ x,
                            const float* __restrict__ mean,
                            float* __restrict__ out,
                            __bf16* __restrict__ A) {
  __shared__ float sbuf0[512 * 16];  // w tile -> softmax in place
  __shared__ float sbuf1[512 * 16];  // reduction scratch -> y tile
  const int b   = blockIdx.y;
  const int t0  = blockIdx.x * 16;
  const int tid = threadIdx.x;
  const int tt  = tid & 15;          // column (invariant across iterations)
  const int s   = tid >> 4;          // 0..15 reduction group
  const size_t xb = (size_t)b * D3_ * T_;

  // Phase 1: load w tile (coalesced over t), emit intensity = exp(w)
#pragma unroll 4
  for (int it = 0; it < 32; ++it) {
    int idx = tid + it * 256;
    int d   = idx >> 4;
    size_t g = xb + (size_t)(D_ + d) * T_ + t0 + tt;
    float wv = x[g];
    sbuf0[idx] = wv;
    out[g] = __expf(wv);
  }
  __syncthreads();

  // Column max over d=512 (partials into sbuf1 rows 0..15, tree reduce)
  {
    float m = -__builtin_inff();
    for (int d = s * 32; d < s * 32 + 32; ++d) m = fmaxf(m, sbuf0[d * 16 + tt]);
    sbuf1[s * 16 + tt] = m;
  }
  __syncthreads();
  for (int st = 8; st >= 1; st >>= 1) {
    if (s < st) sbuf1[s * 16 + tt] = fmaxf(sbuf1[s * 16 + tt], sbuf1[(s + st) * 16 + tt]);
    __syncthreads();
  }
  const float cmax = sbuf1[tt];
  __syncthreads();

  // Column sum of exp(w - max)  (partials in sbuf1 rows 16..31)
  {
    float sum = 0.f;
    for (int d = s * 32; d < s * 32 + 32; ++d) sum += __expf(sbuf0[d * 16 + tt] - cmax);
    sbuf1[(16 + s) * 16 + tt] = sum;
  }
  __syncthreads();
  for (int st = 8; st >= 1; st >>= 1) {
    if (s < st) sbuf1[(16 + s) * 16 + tt] += sbuf1[(16 + s + st) * 16 + tt];
    __syncthreads();
  }
  const float inv = 1.0f / sbuf1[16 * 16 + tt];
  __syncthreads();

  // Softmax in place (each thread owns its own idx set, same tt)
#pragma unroll 4
  for (int it = 0; it < 32; ++it) {
    int idx = tid + it * 256;
    sbuf0[idx] = __expf(sbuf0[idx] - cmax) * inv;
  }
  // Load y tile into sbuf1 (coalesced over t)
#pragma unroll 4
  for (int it = 0; it < 32; ++it) {
    int idx = tid + it * 256;
    int d   = idx >> 4;
    sbuf1[idx] = x[xb + (size_t)d * T_ + t0 + tt];
  }
  __syncthreads();

  // A store: consecutive threads -> consecutive d (coalesced bf16 stores)
#pragma unroll 4
  for (int it = 0; it < 32; ++it) {
    int idx = tid + it * 256;
    int d   = idx & 511;
    int tc  = idx >> 9;   // 0..15
    float a = sbuf0[d * 16 + tc] * (sbuf1[d * 16 + tc] - mean[b * 512 + d]);
    A[((size_t)(b * T_ + t0 + tc)) * 512 + d] = (__bf16)a;
  }
}

// ---------------------------------------------------------------------------
// Kernel 3: rep[t][n] = sum_k A[t][k] * Wt[n][k] + mean[n]; write rep1
// transposed (d-major) and fuse y_trans = x2 - rep1.
// Block 256 = 8 waves; wave tile 16(t) x 64(d) (4 wmma accumulators, A reuse).
// ---------------------------------------------------------------------------
__global__ void k_gemm(const __bf16* __restrict__ A,
                       const __bf16* __restrict__ Wt,
                       const float* __restrict__ mean,
                       const float* __restrict__ x,
                       float* __restrict__ out) {
  const int b    = blockIdx.z;
  const int t0   = blockIdx.x * 128 + (threadIdx.x >> 5) * 16;
  const int d0   = blockIdx.y * 64;
  const int lane = threadIdx.x & 31;
  const int hi   = lane >> 4;   // lane half selects K sub-range
  const int lm   = lane & 15;   // A: row M; B: col N

  // A row for this lane's M (ISA 16-bit A 16x32 layout: lanes hold paired K)
  const __bf16* Arow = A + ((size_t)(b * T_) + t0 + lm) * 512;

  v8f acc0 = {}, acc1 = {}, acc2 = {}, acc3 = {};

  for (int k0 = 0; k0 < 512; k0 += 32) {
    // A fragment: elems 0..15 = K[k0+16*hi .. +15] (contiguous, matches B)
    v8bf alo = *(const v8bf*)(Arow + k0 + 16 * hi);
    v8bf ahi = *(const v8bf*)(Arow + k0 + 16 * hi + 8);
    v16bf a = __builtin_shufflevector(alo, ahi, 0, 1, 2, 3, 4, 5, 6, 7,
                                      8, 9, 10, 11, 12, 13, 14, 15);
    // B fragments: lane half hi holds 16 contiguous K for column N=lm
    const __bf16* Bbase = Wt + (size_t)k0 + 16 * hi + (size_t)(d0 + lm) * 512;
    v16bf b0 = *(const v16bf*)(Bbase + (size_t)0 * 16 * 512);
    v16bf b1 = *(const v16bf*)(Bbase + (size_t)1 * 16 * 512);
    v16bf b2 = *(const v16bf*)(Bbase + (size_t)2 * 16 * 512);
    v16bf b3 = *(const v16bf*)(Bbase + (size_t)3 * 16 * 512);
    acc0 = __builtin_amdgcn_wmma_f32_16x16x32_bf16(false, a, false, b0, (short)0, acc0, false, false);
    acc1 = __builtin_amdgcn_wmma_f32_16x16x32_bf16(false, a, false, b1, (short)0, acc1, false, false);
    acc2 = __builtin_amdgcn_wmma_f32_16x16x32_bf16(false, a, false, b2, (short)0, acc2, false, false);
    acc3 = __builtin_amdgcn_wmma_f32_16x16x32_bf16(false, a, false, b3, (short)0, acc3, false, false);
  }

  // Epilogue: lane owns column d = d0+dt*16+lm, rows t = t0+8*hi+0..7
  v8f accs[4] = {acc0, acc1, acc2, acc3};
#pragma unroll
  for (int dt = 0; dt < 4; ++dt) {
    int d = d0 + dt * 16 + lm;
    float mn = mean[b * 512 + d];
    size_t orow = ((size_t)b * D3_ + d) * T_ + t0 + 8 * hi;            // rep1 row
    size_t trow = ((size_t)b * D3_ + 1024 + d) * T_ + t0 + 8 * hi;     // y_trans row
    size_t xrow = trow;                                                // x2 same channel offset
    v4f r0, r1;
#pragma unroll
    for (int i = 0; i < 4; ++i) { r0[i] = accs[dt][i] + mn; r1[i] = accs[dt][i + 4] + mn; }
    v4f x0 = *(const v4f*)(x + xrow);
    v4f x1 = *(const v4f*)(x + xrow + 4);
    v4f y0, y1;
#pragma unroll
    for (int i = 0; i < 4; ++i) { y0[i] = x0[i] - r0[i]; y1[i] = x1[i] - r1[i]; }
    *(v4f*)(out + orow)     = r0;
    *(v4f*)(out + orow + 4) = r1;
    *(v4f*)(out + trow)     = y0;
    *(v4f*)(out + trow + 4) = y1;
  }
}

// ---------------------------------------------------------------------------
extern "C" void kernel_launch(void* const* d_in, const int* in_sizes, int n_in,
                              void* d_out, int out_size, void* d_ws, size_t ws_size,
                              hipStream_t stream) {
  (void)in_sizes; (void)n_in; (void)out_size; (void)ws_size;
  const float* x = (const float*)d_in[0];   // (32, 1536, 2048) f32
  const float* W = (const float*)d_in[1];   // (512, 512) f32
  float* out = (float*)d_out;               // (32, 1536, 2048) f32

  char* ws = (char*)d_ws;
  __bf16* A    = (__bf16*)ws;                                        // 64 MB
  __bf16* Wt   = (__bf16*)(ws + (size_t)64 * 1024 * 1024);           // 512 KB
  float*  mean = (float*)(ws + (size_t)64 * 1024 * 1024 + 512 * 1024); // 64 KB

  k_wt<<<1024, 256, 0, stream>>>(W, Wt);
  k_mean<<<2048, 256, 0, stream>>>(x, mean);
  k_softmax_A<<<dim3(T_ / 16, B_, 1), 256, 0, stream>>>(x, mean, out, A);
  k_gemm<<<dim3(T_ / 128, D_ / 64, B_), 256, 0, stream>>>(A, Wt, mean, x, out);
}
